// EnhancedLightweightStudent_46720654246434
// MI455X (gfx1250) — matrix operand order repaired and checked by reading, not compiled, not run on hardware
//
#include <hip/hip_runtime.h>

#define Bn 64
#define Tn 512
#define En 256
#define Hn 8
#define DHn 32
#define Sn 513
#define SPn 544
#define Ln 4

typedef _Float16 h16;
typedef __attribute__((ext_vector_type(16))) _Float16 v16h;
typedef __attribute__((ext_vector_type(8)))  _Float16 v8h;
typedef __attribute__((ext_vector_type(8)))  float    v8f;

__device__ __forceinline__ float geluf(float x) {
  return 0.5f * x * (1.0f + erff(x * 0.70710678118654752440f));
}
__device__ __forceinline__ float sigmf(float x) {
  return 1.0f / (1.0f + expf(-x));
}

// ---------------------------------------------------------------------------
// Generic batched WMMA GEMM: C[M,N] = act((A[M,K] * Bt[N,K]^T) * scale + bias)
// A, Bt row-major f16.  One wave computes a 32x64 tile (2 M-frags x 4 N-frags)
// -> 8 v_wmma_f32_16x16x32_f16 per K-step with A reused 4x and B reused 2x.
// Out-of-range rows/cols: fragment pointers are clamped to row/col 0 (valid
// memory); the garbage results only land in D rows/cols that the epilogue
// never stores (WMMA dot products are independent per (m,n), no exceptions),
// so no fragment zeroing is needed in the hot loop.
// Requires: K % 32 == 0, lda/ldb multiples of 32 halves, 32B-aligned bases.
// ACT: 0=none 1=gelu 2=sigmoid. OUTH: write f16. ADD: C += (f32 only).
// ---------------------------------------------------------------------------
template<int ACT, bool OUTH, bool ADD>
__global__ __launch_bounds__(256) void wmma_gemm(
    const h16* __restrict__ A, const h16* __restrict__ Bt,
    const float* __restrict__ bias, const float* __restrict__ scale,
    void* __restrict__ Cv, int M, int N, int K,
    int lda, int ldb, int ldc, long sA, long sB, long sC)
{
  const int lane    = threadIdx.x;
  const int tilesN  = (N + 63) >> 6;            // 64-wide N blocks
  const int tilesM  = (M + 31) >> 5;            // 32-tall M blocks
  const int tile    = blockIdx.x * blockDim.y + threadIdx.y;
  if (tile >= tilesM * tilesN) return;          // wave-uniform exit
  const int row0 = (tile / tilesN) << 5;
  const int col0 = (tile % tilesN) << 6;
  const long zb  = blockIdx.y;

  const h16* Ab = A + (size_t)(zb * sA);
  const h16* Bb = Bt + (size_t)(zb * sB);

  const int hi   = lane >> 4;      // 0 or 1 (lane half)
  const int l15  = lane & 15;
  const int aoff = hi ? 16 : 0;    // A frag: lanes 16-31 hold K=16..31
  const int boff = hi ? 16 : 0;    // B frag: lanes 16-31 hold K=16..31

  const h16* arow[2];
  const h16* brow[4];
#pragma unroll
  for (int mt = 0; mt < 2; ++mt) {
    const int r = row0 + mt * 16 + l15;
    arow[mt] = Ab + (size_t)(r < M ? r : 0) * lda + aoff;
  }
#pragma unroll
  for (int nt = 0; nt < 4; ++nt) {
    const int c = col0 + nt * 16 + l15;
    brow[nt] = Bb + (size_t)(c < N ? c : 0) * ldb + boff;
  }

  v8f acc[2][4] = {};

  for (int k0 = 0; k0 < K; k0 += 32) {
    v16h af[2], bf[4];
#pragma unroll
    for (int mt = 0; mt < 2; ++mt) {
      af[mt] = *(const v16h*)(arow[mt] + k0);       // K = aoff..aoff+15
    }
#pragma unroll
    for (int nt = 0; nt < 4; ++nt) {
      bf[nt] = *(const v16h*)(brow[nt] + k0);       // K = boff..boff+15
    }
#pragma unroll
    for (int mt = 0; mt < 2; ++mt)
#pragma unroll
      for (int nt = 0; nt < 4; ++nt)
        acc[mt][nt] = __builtin_amdgcn_wmma_f32_16x16x32_f16(
            false, af[mt], false, bf[nt], (short)0, acc[mt][nt], false, false);
  }

#pragma unroll
  for (int nt = 0; nt < 4; ++nt) {
    const int col = col0 + nt * 16 + l15;
    if (col < N) {
      const float sc = scale ? scale[col] : 1.0f;
      const float bs = bias ? bias[col] : 0.0f;
#pragma unroll
      for (int mt = 0; mt < 2; ++mt) {
#pragma unroll
        for (int j = 0; j < 8; ++j) {
          const int row = row0 + mt * 16 + j + (hi << 3);
          if (row < M) {
            float v = acc[mt][nt][j] * sc + bs;
            if (ACT == 1) v = geluf(v);
            else if (ACT == 2) v = sigmf(v);
            const size_t idx = (size_t)(zb * sC) + (size_t)row * ldc + col;
            if (OUTH) {
              ((h16*)Cv)[idx] = (h16)v;
            } else {
              float* Cf = (float*)Cv;
              if (ADD) Cf[idx] += v; else Cf[idx] = v;
            }
          }
        }
      }
    }
  }
}

// ---------------------------------------------------------------------------
// LayerNorm over rows (one block per row), optional GELU, f16 or f32 output.
// ---------------------------------------------------------------------------
template<bool GELU, bool OUTH>
__global__ __launch_bounds__(256) void k_ln(const float* __restrict__ in, int cols, int ldin,
    const float* __restrict__ g, const float* __restrict__ bt,
    void* __restrict__ out, int ldout)
{
  const int row = blockIdx.x;
  const float* x = in + (size_t)row * ldin;
  __shared__ float r1[256], r2[256];
  float a = 0.f, b = 0.f;
  for (int c = threadIdx.x; c < cols; c += 256) { float v = x[c]; a += v; b += v * v; }
  r1[threadIdx.x] = a; r2[threadIdx.x] = b;
  __syncthreads();
  for (int o = 128; o > 0; o >>= 1) {
    if (threadIdx.x < o) { r1[threadIdx.x] += r1[threadIdx.x + o]; r2[threadIdx.x] += r2[threadIdx.x + o]; }
    __syncthreads();
  }
  const float mean = r1[0] / cols;
  const float var  = fmaxf(r2[0] / cols - mean * mean, 0.f);
  const float rstd = rsqrtf(var + 1e-5f);
  for (int c = threadIdx.x; c < cols; c += 256) {
    float v = (x[c] - mean) * rstd * g[c] + bt[c];
    if (GELU) v = geluf(v);
    if (OUTH) ((h16*)out)[(size_t)row * ldout + c] = (h16)v;
    else      ((float*)out)[(size_t)row * ldout + c] = v;
  }
}

// Softmax over rows of P [B*SP, SP] (f16, in-place). Rows >= Sn and cols >= Sn -> 0.
__global__ __launch_bounds__(256) void k_softmax(h16* __restrict__ P, float scale)
{
  const size_t row = blockIdx.x;
  const int q = (int)(row % SPn);
  h16* p = P + row * SPn;
  if (q >= Sn) {
    for (int c = threadIdx.x; c < SPn; c += 256) p[c] = (h16)0;
    return;
  }
  __shared__ float red[256];
  float mx = -3.0e38f;
  for (int c = threadIdx.x; c < Sn; c += 256) mx = fmaxf(mx, (float)p[c] * scale);
  red[threadIdx.x] = mx; __syncthreads();
  for (int o = 128; o > 0; o >>= 1) {
    if (threadIdx.x < o) red[threadIdx.x] = fmaxf(red[threadIdx.x], red[threadIdx.x + o]);
    __syncthreads();
  }
  const float m = red[0];
  __syncthreads();
  float sum = 0.f;
  for (int c = threadIdx.x; c < Sn; c += 256) sum += expf((float)p[c] * scale - m);
  red[threadIdx.x] = sum; __syncthreads();
  for (int o = 128; o > 0; o >>= 1) {
    if (threadIdx.x < o) red[threadIdx.x] += red[threadIdx.x + o];
    __syncthreads();
  }
  const float inv = 1.0f / red[0];
  for (int c = threadIdx.x; c < SPn; c += 256) {
    float v = (c < Sn) ? expf((float)p[c] * scale - m) * inv : 0.f;
    p[c] = (h16)v;
  }
}

// BN(eval) fold: s = g * rsqrt(1+eps); bf = conv_b * s + bn_b
__global__ void k_prep_bn(const float* __restrict__ g, const float* __restrict__ bb,
                          const float* __restrict__ cb, float* __restrict__ sc,
                          float* __restrict__ bf, int n)
{
  int i = blockIdx.x * blockDim.x + threadIdx.x;
  if (i >= n) return;
  float s = g[i] * rsqrtf(1.0f + 1e-5f);
  sc[i] = s; bf[i] = cb[i] * s + bb[i];
}

__global__ void k_cast_f2h(const float* __restrict__ s, h16* __restrict__ d, size_t n)
{
  size_t i = (size_t)blockIdx.x * blockDim.x + threadIdx.x;
  if (i < n) d[i] = (h16)s[i];
}

// conv2_w [E,E,3] -> Bt layout [o, dt*E + i]
__global__ void k_conv2w(const float* __restrict__ w, h16* __restrict__ d)
{
  int id = blockIdx.x * blockDim.x + threadIdx.x;
  if (id >= En * En * 3) return;
  const int o = id / 768, r = id % 768, dt = r / En, i = r % En;
  d[id] = (h16)w[o * 768 + i * 3 + dt];
}

// conv1 (Cin=4, k=3) + folded BN + GELU -> x1h [B,T,E] f16
__global__ void k_conv1(const float* __restrict__ acc, const float* __restrict__ w,
                        const float* __restrict__ s1, const float* __restrict__ b1,
                        h16* __restrict__ x1)
{
  size_t id = (size_t)blockIdx.x * blockDim.x + threadIdx.x;
  if (id >= (size_t)Bn * Tn * En) return;
  const int o = (int)(id % En);
  const int t = (int)((id / En) % Tn);
  const int b = (int)(id / ((size_t)En * Tn));
  float sum = 0.f;
#pragma unroll
  for (int dt = 0; dt < 3; ++dt) {
    const int tt = t + dt - 1;
    if (tt < 0 || tt >= Tn) continue;
    const float* ap = acc + ((size_t)b * Tn + tt) * 4;
    const float* wp = w + (size_t)o * 12 + dt;
    sum += ap[0] * wp[0] + ap[1] * wp[3] + ap[2] * wp[6] + ap[3] * wp[9];
  }
  x1[id] = (h16)geluf(sum * s1[o] + b1[o]);
}

// im2col for conv2: xcol[b, t, dt*E + i] = x1[b, t+dt-1, i] (zero-pad)
__global__ void k_im2col(const h16* __restrict__ x1, h16* __restrict__ xc)
{
  size_t id = (size_t)blockIdx.x * blockDim.x + threadIdx.x;
  if (id >= (size_t)Bn * Tn * 768) return;
  const int c = (int)(id % 768);
  const int t = (int)((id / 768) % Tn);
  const int b = (int)(id / ((size_t)768 * Tn));
  const int dt = c / En, i = c % En;
  const int tt = t + dt - 1;
  h16 v = (h16)0;
  if (tt >= 0 && tt < Tn) v = x1[((size_t)b * Tn + tt) * En + i];
  xc[id] = v;
}

__global__ void k_mean_t(const float* __restrict__ x2, float* __restrict__ ca)
{
  int id = blockIdx.x * blockDim.x + threadIdx.x;
  if (id >= Bn * En) return;
  const int e = id % En, b = id / En;
  const float* p = x2 + (size_t)b * Tn * En + e;
  float s = 0.f;
  for (int t = 0; t < Tn; ++t) s += p[(size_t)t * En];
  ca[id] = s * (1.0f / Tn);
}

__global__ void k_mean_e(const float* __restrict__ x2, float* __restrict__ ta)
{
  int id = blockIdx.x * blockDim.x + threadIdx.x;
  if (id >= Bn * Tn) return;
  const float* p = x2 + (size_t)id * En;
  float s = 0.f;
  for (int e = 0; e < En; ++e) s += p[e];
  ta[id] = s * (1.0f / En);
}

__global__ void k_scale_e(float* __restrict__ x2, const float* __restrict__ g1)
{
  size_t id = (size_t)blockIdx.x * blockDim.x + threadIdx.x;
  if (id >= (size_t)Bn * Tn * En) return;
  const int e = (int)(id % En);
  const int b = (int)(id / ((size_t)Tn * En));
  x2[id] *= g1[b * En + e];
}

// xL[b,0,:]=cls ; xL[b,1+t,:] = x2*gate2 + pos_embedding
__global__ void k_assemble(const float* __restrict__ x2, const float* __restrict__ g2,
                           const float* __restrict__ cls, float* __restrict__ xL)
{
  size_t id = (size_t)blockIdx.x * blockDim.x + threadIdx.x;
  if (id >= (size_t)Bn * Sn * En) return;
  const int e = (int)(id % En);
  const int r = (int)((id / En) % Sn);
  const int b = (int)(id / ((size_t)En * Sn));
  float v;
  if (r == 0) v = cls[e];
  else {
    const int t = r - 1;
    const int j2 = (e >> 1) << 1;
    const float ang = (float)t * expf(-(float)j2 * (9.210340371976184f / 256.0f));
    const float pe = (e & 1) ? cosf(ang) : sinf(ang);
    v = x2[((size_t)b * Tn + t) * En + e] * g2[b * Tn + t] + pe;
  }
  xL[id] = v;
}

// qkv [B,S,768] -> Q,K [B,H,S,32] and V^T [B,H,32,SP] (pad cols zeroed)
__global__ void k_pack_qkv(const h16* __restrict__ qkv, h16* __restrict__ Q,
                           h16* __restrict__ Kx, h16* __restrict__ Vt)
{
  size_t id = (size_t)blockIdx.x * blockDim.x + threadIdx.x;
  if (id >= (size_t)Bn * SPn * En) return;
  const int e = (int)(id % En);
  const int s = (int)((id / En) % SPn);
  const int b = (int)(id / ((size_t)En * SPn));
  const int h = e >> 5, d = e & 31;
  const size_t vti = (((size_t)b * Hn + h) * DHn + d) * SPn + s;
  if (s < Sn) {
    const size_t qi = ((size_t)b * Sn + s) * (3 * En);
    const size_t qo = (((size_t)b * Hn + h) * Sn + s) * DHn + d;
    Q[qo]  = qkv[qi + e];
    Kx[qo] = qkv[qi + En + e];
    Vt[vti] = qkv[qi + 2 * En + e];
  } else {
    Vt[vti] = (h16)0;
  }
}

__global__ void k_cls(const float* __restrict__ xL, h16* __restrict__ out)
{
  int id = blockIdx.x * blockDim.x + threadIdx.x;
  if (id >= Bn * En) return;
  const int e = id % En, b = id / En;
  out[id] = (h16)xL[(size_t)b * Sn * En + e];
}

// ---------------------------------------------------------------------------
template<int ACT, bool OUTH, bool ADD>
static void launch_gemm(hipStream_t st, const h16* A, const h16* Bt, const float* bias,
                        const float* scale, void* C, int M, int N, int K,
                        int lda, int ldb, int ldc, long sA, long sB, long sC, int batches)
{
  const int tiles = ((M + 31) / 32) * ((N + 63) / 64);
  dim3 grid((unsigned)((tiles + 7) / 8), (unsigned)batches);
  wmma_gemm<ACT, OUTH, ADD><<<grid, dim3(32, 8), 0, st>>>(
      A, Bt, bias, scale, C, M, N, K, lda, ldb, ldc, sA, sB, sC);
}

static inline unsigned gf(size_t n) { return (unsigned)((n + 255) / 256); }

extern "C" void kernel_launch(void* const* d_in, const int* in_sizes, int n_in,
                              void* d_out, int out_size, void* d_ws, size_t ws_size,
                              hipStream_t stream)
{
  (void)in_sizes; (void)n_in; (void)out_size; (void)ws_size;
  const float* acc_data   = (const float*)d_in[0];
  const float* conv1_w    = (const float*)d_in[1];
  const float* conv1_b    = (const float*)d_in[2];
  const float* bn1_g      = (const float*)d_in[3];
  const float* bn1_b      = (const float*)d_in[4];
  const float* conv2_w    = (const float*)d_in[5];
  const float* conv2_b    = (const float*)d_in[6];
  const float* bn2_g      = (const float*)d_in[7];
  const float* bn2_b      = (const float*)d_in[8];
  const float* sa_w1      = (const float*)d_in[9];
  const float* sa_b1      = (const float*)d_in[10];
  const float* sa_ln_g    = (const float*)d_in[11];
  const float* sa_ln_b    = (const float*)d_in[12];
  const float* sa_w2      = (const float*)d_in[13];
  const float* sa_b2      = (const float*)d_in[14];
  const float* ca_w1      = (const float*)d_in[15];
  const float* ca_b1      = (const float*)d_in[16];
  const float* ca_ln_g    = (const float*)d_in[17];
  const float* ca_ln_b    = (const float*)d_in[18];
  const float* ca_w2      = (const float*)d_in[19];
  const float* ca_b2      = (const float*)d_in[20];
  const float* cls_token  = (const float*)d_in[21];
  const float* n1_g       = (const float*)d_in[22];
  const float* n1_b       = (const float*)d_in[23];
  const float* attn_in_w  = (const float*)d_in[24];
  const float* attn_in_b  = (const float*)d_in[25];
  const float* attn_out_w = (const float*)d_in[26];
  const float* attn_out_b = (const float*)d_in[27];
  const float* n2_g       = (const float*)d_in[28];
  const float* n2_b       = (const float*)d_in[29];
  const float* mlp_w1     = (const float*)d_in[30];
  const float* mlp_b1     = (const float*)d_in[31];
  const float* mlp_w2     = (const float*)d_in[32];
  const float* mlp_b2     = (const float*)d_in[33];
  const float* fus_w      = (const float*)d_in[34];
  const float* fus_b      = (const float*)d_in[35];
  const float* fus_ln_g   = (const float*)d_in[36];
  const float* fus_ln_b   = (const float*)d_in[37];
  const float* clf_ln_g   = (const float*)d_in[38];
  const float* clf_ln_b   = (const float*)d_in[39];
  const float* clf_w1     = (const float*)d_in[40];
  const float* clf_b1     = (const float*)d_in[41];
  const float* clf_w2     = (const float*)d_in[42];
  const float* clf_b2     = (const float*)d_in[43];
  float* out = (float*)d_out;

  // ---- bump allocator on workspace (needs ~380 MB) ----
  char* base = (char*)d_ws;
  size_t off = 0;
  auto alloc = [&](size_t bytes) -> void* {
    void* p = base + off;
    off = (off + bytes + 255) & ~(size_t)255;
    return p;
  };

  float* s1v = (float*)alloc(En * 4);
  float* b1v = (float*)alloc(En * 4);
  float* s2v = (float*)alloc(En * 4);
  float* b2v = (float*)alloc(En * 4);

  h16* w_conv2t  = (h16*)alloc((size_t)En * 768 * 2);
  h16* w_sa1     = (h16*)alloc((size_t)32 * En * 2);
  h16* w_sa2     = (h16*)alloc((size_t)En * 32 * 2);
  h16* w_ca1     = (h16*)alloc((size_t)64 * Tn * 2);
  h16* w_ca2     = (h16*)alloc((size_t)Tn * 64 * 2);
  h16* w_ain     = (h16*)alloc((size_t)Ln * 768 * En * 2);
  h16* w_aout    = (h16*)alloc((size_t)Ln * En * En * 2);
  h16* w_m1      = (h16*)alloc((size_t)Ln * 1024 * En * 2);
  h16* w_m2      = (h16*)alloc((size_t)Ln * En * 1024 * 2);
  h16* w_fus     = (h16*)alloc((size_t)Ln * En * En * 2);
  h16* w_c1      = (h16*)alloc((size_t)En * 1024 * 2);
  h16* w_c2      = (h16*)alloc((size_t)2 * En * 2);

  h16*   x1h   = (h16*)alloc((size_t)Bn * Tn * En * 2);
  h16*   xcol  = (h16*)alloc((size_t)Bn * Tn * 768 * 2);
  float* x2    = (float*)alloc((size_t)Bn * Tn * En * 4);
  float* camn  = (float*)alloc((size_t)Bn * En * 4);
  h16*   ca_h  = (h16*)alloc((size_t)Bn * En * 2);
  float* t_se1 = (float*)alloc((size_t)Bn * 32 * 4);
  h16*   se1_h = (h16*)alloc((size_t)Bn * 32 * 2);
  float* gate1 = (float*)alloc((size_t)Bn * En * 4);
  float* tamn  = (float*)alloc((size_t)Bn * Tn * 4);
  h16*   ta_h  = (h16*)alloc((size_t)Bn * Tn * 2);
  float* t_se2 = (float*)alloc((size_t)Bn * 64 * 4);
  h16*   se2_h = (h16*)alloc((size_t)Bn * 64 * 2);
  float* gate2 = (float*)alloc((size_t)Bn * Tn * 4);
  float* xL    = (float*)alloc((size_t)Bn * Sn * En * 4);
  h16*   xn_h  = (h16*)alloc((size_t)Bn * Sn * En * 2);
  h16*   qkv_h = (h16*)alloc((size_t)Bn * Sn * 768 * 2);
  h16*   Qh    = (h16*)alloc((size_t)Bn * Hn * Sn * DHn * 2);
  h16*   Kh    = (h16*)alloc((size_t)Bn * Hn * Sn * DHn * 2);
  h16*   Vt    = (h16*)alloc((size_t)Bn * Hn * DHn * SPn * 2);
  h16*   Ph    = (h16*)alloc((size_t)Bn * SPn * SPn * 2);
  h16*   ao_h  = (h16*)alloc((size_t)Bn * Sn * En * 2);
  h16*   mlph  = (h16*)alloc((size_t)Bn * Sn * 1024 * 2);
  h16*   clsr  = (h16*)alloc((size_t)Bn * En * 2);
  float* ftmp  = (float*)alloc((size_t)Bn * En * 4);
  float* cat   = (float*)alloc((size_t)Bn * 1024 * 4);
  h16*   cat_h = (h16*)alloc((size_t)Bn * 1024 * 2);
  h16*   c1_h  = (h16*)alloc((size_t)Bn * En * 2);

  // ---- weight prep ----
  k_prep_bn<<<1, 256, 0, stream>>>(bn1_g, bn1_b, conv1_b, s1v, b1v, En);
  k_prep_bn<<<1, 256, 0, stream>>>(bn2_g, bn2_b, conv2_b, s2v, b2v, En);
  k_conv2w<<<gf((size_t)En * 768), 256, 0, stream>>>(conv2_w, w_conv2t);
  k_cast_f2h<<<gf((size_t)32 * En), 256, 0, stream>>>(sa_w1, w_sa1, (size_t)32 * En);
  k_cast_f2h<<<gf((size_t)En * 32), 256, 0, stream>>>(sa_w2, w_sa2, (size_t)En * 32);
  k_cast_f2h<<<gf((size_t)64 * Tn), 256, 0, stream>>>(ca_w1, w_ca1, (size_t)64 * Tn);
  k_cast_f2h<<<gf((size_t)Tn * 64), 256, 0, stream>>>(ca_w2, w_ca2, (size_t)Tn * 64);
  k_cast_f2h<<<gf((size_t)Ln * 768 * En), 256, 0, stream>>>(attn_in_w, w_ain, (size_t)Ln * 768 * En);
  k_cast_f2h<<<gf((size_t)Ln * En * En), 256, 0, stream>>>(attn_out_w, w_aout, (size_t)Ln * En * En);
  k_cast_f2h<<<gf((size_t)Ln * 1024 * En), 256, 0, stream>>>(mlp_w1, w_m1, (size_t)Ln * 1024 * En);
  k_cast_f2h<<<gf((size_t)Ln * En * 1024), 256, 0, stream>>>(mlp_w2, w_m2, (size_t)Ln * En * 1024);
  k_cast_f2h<<<gf((size_t)Ln * En * En), 256, 0, stream>>>(fus_w, w_fus, (size_t)Ln * En * En);
  k_cast_f2h<<<gf((size_t)En * 1024), 256, 0, stream>>>(clf_w1, w_c1, (size_t)En * 1024);
  k_cast_f2h<<<gf((size_t)2 * En), 256, 0, stream>>>(clf_w2, w_c2, (size_t)2 * En);

  // ---- conv stem ----
  k_conv1<<<gf((size_t)Bn * Tn * En), 256, 0, stream>>>(acc_data, conv1_w, s1v, b1v, x1h);
  k_im2col<<<gf((size_t)Bn * Tn * 768), 256, 0, stream>>>(x1h, xcol);
  // conv2 as GEMM, BN folded, GELU. x2 [B,T,E] f32
  launch_gemm<1, false, false>(stream, xcol, w_conv2t, b2v, s2v, x2,
                               Tn, En, 768, 768, 768, En,
                               (long)Tn * 768, 0, (long)Tn * En, Bn);

  // ---- SE gates ----
  k_mean_t<<<gf((size_t)Bn * En), 256, 0, stream>>>(x2, camn);
  k_cast_f2h<<<gf((size_t)Bn * En), 256, 0, stream>>>(camn, ca_h, (size_t)Bn * En);
  launch_gemm<0, false, false>(stream, ca_h, w_sa1, sa_b1, nullptr, t_se1,
                               Bn, 32, En, En, En, 32, 0, 0, 0, 1);
  k_ln<true, true><<<Bn, 256, 0, stream>>>(t_se1, 32, 32, sa_ln_g, sa_ln_b, se1_h, 32);
  launch_gemm<2, false, false>(stream, se1_h, w_sa2, sa_b2, nullptr, gate1,
                               Bn, En, 32, 32, 32, En, 0, 0, 0, 1);
  k_scale_e<<<gf((size_t)Bn * Tn * En), 256, 0, stream>>>(x2, gate1);
  k_mean_e<<<gf((size_t)Bn * Tn), 256, 0, stream>>>(x2, tamn);
  k_cast_f2h<<<gf((size_t)Bn * Tn), 256, 0, stream>>>(tamn, ta_h, (size_t)Bn * Tn);
  launch_gemm<0, false, false>(stream, ta_h, w_ca1, ca_b1, nullptr, t_se2,
                               Bn, 64, Tn, Tn, Tn, 64, 0, 0, 0, 1);
  k_ln<true, true><<<Bn, 256, 0, stream>>>(t_se2, 64, 64, ca_ln_g, ca_ln_b, se2_h, 64);
  launch_gemm<2, false, false>(stream, se2_h, w_ca2, ca_b2, nullptr, gate2,
                               Bn, Tn, 64, 64, 64, Tn, 0, 0, 0, 1);
  k_assemble<<<gf((size_t)Bn * Sn * En), 256, 0, stream>>>(x2, gate2, cls_token, xL);

  // ---- transformer layers ----
  const float attn_scale = 0.17677669529663688f; // 1/sqrt(32)
  for (int i = 0; i < Ln; ++i) {
    // pre-norm MHA
    k_ln<false, true><<<Bn * Sn, 256, 0, stream>>>(xL, En, En, n1_g + i * En, n1_b + i * En, xn_h, En);
    launch_gemm<0, true, false>(stream, xn_h, w_ain + (size_t)i * 768 * En,
                                attn_in_b + i * 768, nullptr, qkv_h,
                                Sn, 768, En, En, En, 768,
                                (long)Sn * En, 0, (long)Sn * 768, Bn);
    k_pack_qkv<<<gf((size_t)Bn * SPn * En), 256, 0, stream>>>(qkv_h, Qh, Kh, Vt);
    for (int h = 0; h < Hn; ++h) {
      // scores = Q K^T  (per-head, batched over B), f16 out into padded P
      launch_gemm<0, true, false>(stream,
          Qh + (size_t)h * Sn * DHn, Kh + (size_t)h * Sn * DHn, nullptr, nullptr, Ph,
          Sn, Sn, DHn, DHn, DHn, SPn,
          (long)Hn * Sn * DHn, (long)Hn * Sn * DHn, (long)SPn * SPn, Bn);
      k_softmax<<<Bn * SPn, 256, 0, stream>>>(Ph, attn_scale);
      // O = P V  -> ao_h column slice for this head
      launch_gemm<0, true, false>(stream,
          Ph, Vt + (size_t)h * DHn * SPn, nullptr, nullptr, ao_h + (size_t)h * DHn,
          Sn, DHn, SPn, SPn, SPn, En,
          (long)SPn * SPn, (long)Hn * DHn * SPn, (long)Sn * En, Bn);
    }
    // out-proj + residual
    launch_gemm<0, false, true>(stream, ao_h, w_aout + (size_t)i * En * En,
                                attn_out_b + i * En, nullptr, xL,
                                Sn, En, En, En, En, En,
                                (long)Sn * En, 0, (long)Sn * En, Bn);
    // pre-norm MLP
    k_ln<false, true><<<Bn * Sn, 256, 0, stream>>>(xL, En, En, n2_g + i * En, n2_b + i * En, xn_h, En);
    launch_gemm<1, true, false>(stream, xn_h, w_m1 + (size_t)i * 1024 * En,
                                mlp_b1 + i * 1024, nullptr, mlph,
                                Sn, 1024, En, En, En, 1024,
                                (long)Sn * En, 0, (long)Sn * 1024, Bn);
    launch_gemm<0, false, true>(stream, mlph, w_m2 + (size_t)i * En * 1024,
                                mlp_b2 + i * En, nullptr, xL,
                                Sn, En, 1024, 1024, 1024, En,
                                (long)Sn * 1024, 0, (long)Sn * En, Bn);
    // fusion head on CLS token
    k_cls<<<gf((size_t)Bn * En), 256, 0, stream>>>(xL, clsr);
    launch_gemm<0, false, false>(stream, clsr, w_fus + (size_t)i * En * En,
                                 fus_b + i * En, nullptr, ftmp,
                                 Bn, En, En, En, En, En, 0, 0, 0, 1);
    k_ln<true, false><<<Bn, 256, 0, stream>>>(ftmp, En, En, fus_ln_g + i * En, fus_ln_b + i * En,
                                              cat + (size_t)i * En, Ln * En);
  }

  // ---- classifier ----
  k_ln<false, true><<<Bn, 256, 0, stream>>>(cat, Ln * En, Ln * En, clf_ln_g, clf_ln_b, cat_h, Ln * En);
  launch_gemm<1, true, false>(stream, cat_h, w_c1, clf_b1, nullptr, c1_h,
                              Bn, En, Ln * En, Ln * En, Ln * En, En, 0, 0, 0, 1);
  launch_gemm<0, false, false>(stream, c1_h, w_c2, clf_b2, nullptr, out,
                               Bn, 2, En, En, En, 2, 0, 0, 0, 1);
}
